// GraphTransformer_38165079392510
// MI455X (gfx1250) — matrix operand order repaired and checked by reading, not compiled, not run on hardware
//
#include <hip/hip_runtime.h>
#include <hip/hip_bf16.h>

typedef __attribute__((ext_vector_type(16))) _Float16 v16h;
typedef __attribute__((ext_vector_type(8)))  float    v8f;

#define DD 64
#define HH 4
#define LL 2
#define HD 256   // HH*DD

// ---- ordered-uint encoding for float atomicMax (handles negatives) ----
__device__ __forceinline__ unsigned f2ord(float f){
  unsigned u = __float_as_uint(f);
  return (u & 0x80000000u) ? ~u : (u | 0x80000000u);
}
__device__ __forceinline__ float ord2f(unsigned u){
  return (u & 0x80000000u) ? __uint_as_float(u & 0x7fffffffu)
                           : __uint_as_float(~u);
}

// ---- convert weight [K,Ncols] f32 (row-major) -> transposed [Ncols,K] f16 ----
__global__ void k_convert_wt(const float* __restrict__ W, _Float16* __restrict__ Wt,
                             int K, int Ncols){
  int t = blockIdx.x * blockDim.x + threadIdx.x;
  if (t >= K * Ncols) return;
  int n = t / K;
  int k = t - n * K;
  Wt[t] = (_Float16)W[(size_t)k * Ncols + n];
}

__global__ void k_bias_enc(const float* __restrict__ bf, const float* __restrict__ bc,
                           float* __restrict__ be){
  int d = threadIdx.x;
  if (d < DD) be[d] = bf[d] + bc[d];
}

// ---- WMMA GEMM: C[M,NCOLS] = A[M,K](f32, optional row gather) * Bt^T + bias ----
// Bt = weights pre-transposed [NCOLS, K] f16.  One wave owns a 16-row x 64-col
// group (4 column tiles): A fragments converted to f16 ONCE, reused across the
// 4 tiles; all loops fully unroll.  K, NCOLS compile-time -> immediate-offset
// stores, no per-element address math.  Tail tiles use a wave-uniform branch.
template<int K, int NCOLS>
__global__ void k_wmma_gemm_t(const float* __restrict__ A, const int* __restrict__ rowIdx,
                              const _Float16* __restrict__ Bt, const float* __restrict__ bias,
                              float* __restrict__ C, int M){
  constexpr int KT  = K / 32;
  constexpr int WPR = NCOLS / 64;            // waves per row-stripe
  int wave = (int)((blockIdx.x * blockDim.x + threadIdx.x) >> 5);
  int lane = (int)(threadIdx.x & 31);
  int mt   = (M + 15) >> 4;
  if (wave >= mt * WPR) return;              // wave-uniform: EXEC all-ones at WMMA
  int rt = wave / WPR;
  int cg = wave - rt * WPR;                  // column group index

  // A operand: lane = row (mod 16); K contiguous per lane half (k = 16h + s), matching B
  int ar   = rt * 16 + (lane & 15);
  int arc  = ar < M ? ar : (M - 1);
  int grow = rowIdx ? rowIdx[arc] : arc;
  const float* arow = A + (size_t)grow * K;
  const int ka = (lane >= 16) ? 16 : 0;      // A K sub-offset for this lane half
  const int kb = (lane >= 16) ? 16 : 0;      // B K sub-offset for this lane half

  // ---- load + convert A fragments once, keep resident in VGPRs ----
  v16h afrag[KT];
  #pragma unroll
  for (int kt = 0; kt < KT; ++kt){
    const float4* pa0 = reinterpret_cast<const float4*>(arow + kt*32 + ka);
    const float4* pa1 = reinterpret_cast<const float4*>(arow + kt*32 + ka + 8);
    float4 f0 = pa0[0], f1 = pa0[1];
    float4 f2 = pa1[0], f3 = pa1[1];
    v16h a;
    a[0]=(_Float16)f0.x;  a[1]=(_Float16)f0.y;  a[2]=(_Float16)f0.z;  a[3]=(_Float16)f0.w;
    a[4]=(_Float16)f1.x;  a[5]=(_Float16)f1.y;  a[6]=(_Float16)f1.z;  a[7]=(_Float16)f1.w;
    a[8]=(_Float16)f2.x;  a[9]=(_Float16)f2.y;  a[10]=(_Float16)f2.z; a[11]=(_Float16)f2.w;
    a[12]=(_Float16)f3.x; a[13]=(_Float16)f3.y; a[14]=(_Float16)f3.z; a[15]=(_Float16)f3.w;
    afrag[kt] = a;
  }

  int rbase = rt * 16 + ((lane >> 4) << 3);  // lanes 16-31 hold rows +8
  bool full = (rt * 16 + 16) <= M;           // wave-uniform

  #pragma unroll
  for (int i = 0; i < 4; ++i){               // 4 column tiles, reusing A fragments
    int colg = (cg << 6) + (i << 4) + (lane & 15);
    const _Float16* brow = Bt + (size_t)colg * K + kb;
    v8f acc = {};
    #pragma unroll
    for (int kt = 0; kt < KT; ++kt){
      v16h b = *reinterpret_cast<const v16h*>(brow + kt*32);
      acc = __builtin_amdgcn_wmma_f32_16x16x32_f16(false, afrag[kt], false, b,
                                                   (short)0, acc, false, false);
    }
    float bv = bias[colg];
    float* crow = C + (size_t)rbase * NCOLS + colg;
    if (full){
      #pragma unroll
      for (int j = 0; j < 8; ++j)
        crow[(size_t)j * NCOLS] = acc[j] + bv;   // immediate-offset stores
    } else {
      #pragma unroll
      for (int j = 0; j < 8; ++j)
        if (rbase + j < M) crow[(size_t)j * NCOLS] = acc[j] + bv;
    }
  }
}

// ---- x += central @ Wc  (rank-2 update) ----
__global__ void k_centrality(float* __restrict__ x, const int* __restrict__ nodes,
                             const float* __restrict__ bet, const float* __restrict__ clo,
                             const float* __restrict__ Wc, int Nn){
  int t = blockIdx.x * blockDim.x + threadIdx.x;
  if (t >= Nn * DD) return;
  int n = t >> 6, d = t & 63;
  int node = nodes[n];
  x[t] += bet[node] * Wc[d] + clo[node] * Wc[DD + d];
}

__global__ void k_init_mz(unsigned* __restrict__ m, float* __restrict__ z, int n){
  int t = blockIdx.x * blockDim.x + threadIdx.x;
  if (t >= n) return;
  m[t] = 0x007FFFFFu;   // f2ord(-inf)
  z[t] = 0.0f;
}

__global__ void k_zero(float* __restrict__ p, int n){
  int t = blockIdx.x * blockDim.x + threadIdx.x;
  if (t < n) p[t] = 0.0f;
}

// ---- per-edge, per-head QK dot; segment max via atomicMax(ordered) ----
__global__ void k_scores(const float* __restrict__ Q, const float* __restrict__ Km,
                         const int* __restrict__ src, const int* __restrict__ tgt,
                         const float* __restrict__ sgn, float* __restrict__ sc,
                         unsigned* __restrict__ m, int E){
  int t = blockIdx.x * blockDim.x + threadIdx.x;
  if (t >= E * HH) return;
  int e = t >> 2, h = t & 3;
  int s = src[e], g = tgt[e];
  const float4* q = reinterpret_cast<const float4*>(Q  + (size_t)s * HD + h * DD);
  const float4* k = reinterpret_cast<const float4*>(Km + (size_t)g * HD + h * DD);
  float acc = 0.0f;
  #pragma unroll
  for (int i = 0; i < DD/4; ++i){
    float4 a = q[i], b = k[i];
    acc += a.x*b.x + a.y*b.y + a.z*b.z + a.w*b.w;
  }
  acc *= 0.125f * sgn[s];           // 1/sqrt(64) * sign[src]
  sc[t] = acc;
  atomicMax(m + (size_t)s * HH + h, f2ord(acc));
}

// ---- e = exp(s - max); z += e ----
__global__ void k_expz(const int* __restrict__ src, float* __restrict__ sc,
                       const unsigned* __restrict__ m, float* __restrict__ z, int E){
  int t = blockIdx.x * blockDim.x + threadIdx.x;
  if (t >= E * HH) return;
  int e = t >> 2, h = t & 3;
  int s = src[e];
  float ev = __expf(sc[t] - ord2f(m[(size_t)s * HH + h]));
  sc[t] = ev;
  atomicAdd(z + (size_t)s * HH + h, ev);
}

// ---- agg[src] += V[tgt] * attn ----
__global__ void k_agg(const float* __restrict__ V, const float* __restrict__ sc,
                      const float* __restrict__ z, const int* __restrict__ src,
                      const int* __restrict__ tgt, float* __restrict__ agg, int E){
  int t = blockIdx.x * blockDim.x + threadIdx.x;
  if (t >= E * DD) return;
  int e = t >> 6, d = t & 63;
  int s = src[e], g = tgt[e];
  #pragma unroll
  for (int h = 0; h < HH; ++h){
    float attn = sc[(size_t)e * HH + h] / (z[(size_t)s * HH + h] + 1e-10f);
    atomicAdd(agg + (size_t)s * HD + h * DD + d,
              V[(size_t)g * HD + h * DD + d] * attn);
  }
}

// ---- x = LayerNorm(x + out) ; one wave32 per node, 2 elems/lane ----
__global__ void k_ln(float* __restrict__ x, const float* __restrict__ ob,
                     const float* __restrict__ g, const float* __restrict__ b, int Nn){
  int wave = (int)((blockIdx.x * blockDim.x + threadIdx.x) >> 5);
  int lane = (int)(threadIdx.x & 31);
  if (wave >= Nn) return;
  size_t base = (size_t)wave * DD;
  float y0 = x[base + lane]      + ob[base + lane];
  float y1 = x[base + lane + 32] + ob[base + lane + 32];
  float s = y0 + y1;
  #pragma unroll
  for (int off = 16; off; off >>= 1) s += __shfl_xor(s, off, 32);
  float mu = s * (1.0f / DD);
  float d0 = y0 - mu, d1 = y1 - mu;
  float v = d0*d0 + d1*d1;
  #pragma unroll
  for (int off = 16; off; off >>= 1) v += __shfl_xor(v, off, 32);
  float inv = rsqrtf(v * (1.0f / DD) + 1e-5f);
  x[base + lane]      = d0 * inv * g[lane]      + b[lane];
  x[base + lane + 32] = d1 * inv * g[lane + 32] + b[lane + 32];
}

extern "C" void kernel_launch(void* const* d_in, const int* in_sizes, int n_in,
                              void* d_out, int out_size, void* d_ws, size_t ws_size,
                              hipStream_t stream) {
  const int*   nodes = (const int*)  d_in[0];
  const float* feat  = (const float*)d_in[1];
  const float* bet   = (const float*)d_in[2];
  const float* clo   = (const float*)d_in[3];
  const float* sgn   = (const float*)d_in[4];
  const int*   src   = (const int*)  d_in[5];
  const int*   tgt   = (const int*)  d_in[6];
  const float* Wf    = (const float*)d_in[7];
  const float* bf    = (const float*)d_in[8];
  const float* Wc    = (const float*)d_in[9];
  const float* bc    = (const float*)d_in[10];
  const float* Wq    = (const float*)d_in[11];
  const float* bq    = (const float*)d_in[12];
  const float* Wk    = (const float*)d_in[13];
  const float* bk    = (const float*)d_in[14];
  const float* Wv    = (const float*)d_in[15];
  const float* bv    = (const float*)d_in[16];
  const float* Wo    = (const float*)d_in[17];
  const float* bo    = (const float*)d_in[18];
  const float* lng   = (const float*)d_in[19];
  const float* lnb   = (const float*)d_in[20];

  const int N = in_sizes[0];
  const int E = in_sizes[5];

  // ---- workspace carving (256B aligned) ----
  char* wsb = (char*)d_ws;
  size_t cur = 0;
  auto carve = [&](size_t bytes) -> void* {
    void* p = wsb + cur;
    cur += (bytes + 255) & ~(size_t)255;
    return p;
  };
  _Float16* Wh       = (_Float16*)carve(sizeof(_Float16) * (4096 + (size_t)LL * 4 * 16384));
  float*    bias_enc = (float*)   carve(sizeof(float) * DD);
  float*    Q        = (float*)   carve(sizeof(float) * (size_t)N * HD);
  float*    Km       = (float*)   carve(sizeof(float) * (size_t)N * HD);
  float*    V        = (float*)   carve(sizeof(float) * (size_t)N * HD);
  float*    sc       = (float*)   carve(sizeof(float) * (size_t)E * HH);
  unsigned* mo       = (unsigned*)carve(sizeof(unsigned) * (size_t)N * HH);
  float*    z        = (float*)   carve(sizeof(float) * (size_t)N * HH);
  float*    agg = Q;   // Q dead after k_scores
  float*    ob  = Km;  // K dead after k_scores
  float*    x   = (float*)d_out;

  _Float16* WfT = Wh;
  auto WT = [&](int l, int which) -> _Float16* {
    return Wh + 4096 + (size_t)l * 4 * 16384 + (size_t)which * 16384;
  };

  const int T = 256;  // 8 waves/block on wave32
  const int mt = (N + 15) / 16;

  // ---- weight conversion (f32 -> transposed f16) ----
  k_convert_wt<<<(64*64 + T-1)/T, T, 0, stream>>>(Wf, WfT, 64, 64);
  for (int l = 0; l < LL; ++l){
    k_convert_wt<<<(64*256 + T-1)/T, T, 0, stream>>>(Wq + (size_t)l*64*256, WT(l,0), 64, 256);
    k_convert_wt<<<(64*256 + T-1)/T, T, 0, stream>>>(Wk + (size_t)l*64*256, WT(l,1), 64, 256);
    k_convert_wt<<<(64*256 + T-1)/T, T, 0, stream>>>(Wv + (size_t)l*64*256, WT(l,2), 64, 256);
    k_convert_wt<<<(256*64 + T-1)/T, T, 0, stream>>>(Wo + (size_t)l*256*64, WT(l,3), 256, 64);
  }
  k_bias_enc<<<1, 64, 0, stream>>>(bf, bc, bias_enc);

  // ---- encoder: x = feat[nodes] @ Wf + (bf+bc), then += central @ Wc ----
  k_wmma_gemm_t<64,64><<<(mt + 7)/8, T, 0, stream>>>(feat, nodes, WfT, bias_enc, x, N);
  k_centrality<<<(unsigned)(((size_t)N*DD + T-1)/T), T, 0, stream>>>(x, nodes, bet, clo, Wc, N);

  // ---- transformer layers ----
  for (int l = 0; l < LL; ++l){
    int wqkv = mt * (HD / 64);
    k_wmma_gemm_t<64,256><<<(wqkv + 7)/8, T, 0, stream>>>(x, nullptr, WT(l,0), bq + (size_t)l*HD, Q,  N);
    k_wmma_gemm_t<64,256><<<(wqkv + 7)/8, T, 0, stream>>>(x, nullptr, WT(l,1), bk + (size_t)l*HD, Km, N);
    k_wmma_gemm_t<64,256><<<(wqkv + 7)/8, T, 0, stream>>>(x, nullptr, WT(l,2), bv + (size_t)l*HD, V,  N);

    k_init_mz<<<(N*HH + T-1)/T, T, 0, stream>>>(mo, z, N*HH);
    k_scores<<<(E*HH + T-1)/T, T, 0, stream>>>(Q, Km, src, tgt, sgn, sc, mo, E);
    k_expz  <<<(E*HH + T-1)/T, T, 0, stream>>>(src, sc, mo, z, E);
    k_zero  <<<(unsigned)(((size_t)N*HD + T-1)/T), T, 0, stream>>>(agg, N*HD);
    k_agg   <<<(unsigned)(((size_t)E*DD + T-1)/T), T, 0, stream>>>(V, sc, z, src, tgt, agg, E);

    k_wmma_gemm_t<256,64><<<(mt + 7)/8, T, 0, stream>>>(agg, nullptr, WT(l,3), bo + (size_t)l*64, ob, N);
    k_ln<<<(N + 7)/8, T, 0, stream>>>(x, ob, lng + (size_t)l*DD, lnb + (size_t)l*DD, N);
  }

  (void)n_in; (void)out_size; (void)ws_size;
}
